// AFTFull_78434692759653
// MI455X (gfx1250) — compile-verified, matrix operand order checked by reading, not run on hardware
//
#include <hip/hip_runtime.h>
#include <hip/hip_bf16.h>
#include <cstdint>

#define USE_TDM 1

typedef __bf16 bf16_t;
typedef bf16_t v16bf __attribute__((ext_vector_type(16)));
typedef float  v8f   __attribute__((ext_vector_type(8)));
typedef uint32_t u32x4 __attribute__((ext_vector_type(4)));
typedef uint32_t u32x8 __attribute__((ext_vector_type(8)));

static __device__ __forceinline__ bf16_t f2b(float f) {
    return __builtin_bit_cast(bf16_t, __float2bfloat16(f));
}

#if USE_TDM
// Issue one Tensor-Data-Mover 2D tile load: 128 rows x 32 bf16 elements,
// global (row-major, row_stride elements) -> LDS (packed 128x32).
// D# layout per CDNA5 ISA.
static __device__ __forceinline__ void tdm_load_tile(const bf16_t* gbase,
                                                     uint32_t lds_off,
                                                     uint32_t row_stride_elems) {
    const uint64_t ga = (uint64_t)(uintptr_t)gbase;
    u32x4 g0;
    g0[0] = 1u;                                           // count=1, user desc
    g0[1] = lds_off;                                      // lds_addr (bytes)
    g0[2] = (uint32_t)ga;                                 // global_addr[31:0]
    g0[3] = (uint32_t)((ga >> 32) & 0x01FFFFFFu)          // global_addr[56:32]
          | (2u << 30);                                   // type = 2 (image)
    u32x8 g1;
    g1[0] = (1u << 16);                                   // data_size=1 -> 2 bytes
    g1[1] = (32u << 16);                                  // tensor_dim0 lo = 32
    g1[2] = (128u << 16);                                 // td0 hi=0 | tensor_dim1 lo = 128
    g1[3] = (32u << 16);                                  // td1 hi=0 | tile_dim0 = 32
    g1[4] = 128u;                                         // tile_dim1 = 128, tile_dim2 = 0
    g1[5] = row_stride_elems;                             // tensor_dim0_stride[31:0]
    g1[6] = 0u;                                           // stride hi / dim1_stride lo
    g1[7] = 0u;
    asm volatile("tensor_load_to_lds %0, %1"
                 :
                 : "s"(g0), "s"(g1)
                 : "memory");
}
#endif

// ---------------------------------------------------------------------------
// bf16 GEMM, f32 accumulate:  C[M,N] = A[M,K] @ Bt[N,K]^T (+ bias[N])
// A and Bt row-major bf16 (Bt is the pre-transposed B operand); C row-major
// f32. M,N multiples of 128; K multiple of 32.
// Block: 256 threads = 8 waves; block tile 128x128; wave tile 32x64 (2x4 WMMA).
// Tiles staged into double-buffered LDS by the Tensor Data Mover (wave 0),
// pipelined one tile ahead; synchronized with TENSORcnt + workgroup barrier.
// ---------------------------------------------------------------------------
__global__ __launch_bounds__(256) void gemm_bf16f32(
    const bf16_t* __restrict__ A, int lda, long long strideA,
    const bf16_t* __restrict__ Bt, int ldb, long long strideB,
    float* __restrict__ C, int ldc, long long strideC,
    const float* __restrict__ bias, int K)
{
    __shared__ __align__(128) bf16_t sA[2][128 * 32];
    __shared__ __align__(128) bf16_t sB[2][128 * 32];

    const int tid  = threadIdx.x;
    const int lane = tid & 31;
    const int wave = tid >> 5;
    const int wm   = wave >> 1;   // 0..3 (M)
    const int wn   = wave & 1;    // 0..1 (N)

    const int m0 = blockIdx.y * 128;
    const int n0 = blockIdx.x * 128;
    A  += (long long)blockIdx.z * strideA + (long long)m0 * lda;
    Bt += (long long)blockIdx.z * strideB + (long long)n0 * ldb;
    C  += (long long)blockIdx.z * strideC;

    v8f acc[2][4];
#pragma unroll
    for (int i = 0; i < 2; ++i)
#pragma unroll
        for (int j = 0; j < 4; ++j)
            acc[i][j] = (v8f){0.f, 0.f, 0.f, 0.f, 0.f, 0.f, 0.f, 0.f};

    // Fragment k order used here for A and B: lanes 0-15 -> K 0..15, lanes 16-31 -> K 16..31
    const int fr = lane & 15;
    const int kh = (lane >> 4) * 16;
    const int nk = K / 32;

#if USE_TDM
    const uint32_t ldsA[2] = { (uint32_t)(uintptr_t)&sA[0][0],
                               (uint32_t)(uintptr_t)&sA[1][0] };
    const uint32_t ldsB[2] = { (uint32_t)(uintptr_t)&sB[0][0],
                               (uint32_t)(uintptr_t)&sB[1][0] };
    if (wave == 0) {
        tdm_load_tile(A,  ldsA[0], (uint32_t)lda);
        tdm_load_tile(Bt, ldsB[0], (uint32_t)ldb);
    }
#endif

    for (int kt = 0; kt < nk; ++kt) {
        const int buf = kt & 1;
#if USE_TDM
        if (wave == 0) {
            if (kt + 1 < nk) {
                // prefetch next tile into the other buffer, then wait for the
                // current tile (TDM ops from one wave complete in order).
                tdm_load_tile(A  + (kt + 1) * 32, ldsA[buf ^ 1], (uint32_t)lda);
                tdm_load_tile(Bt + (kt + 1) * 32, ldsB[buf ^ 1], (uint32_t)ldb);
                __builtin_amdgcn_s_wait_tensorcnt(2);
            } else {
                __builtin_amdgcn_s_wait_tensorcnt(0);
            }
        }
        __syncthreads();
#else
        const int srow  = tid >> 1;           // 0..127
        const int shalf = (tid & 1) * 16;     // 0 or 16
        v16bf av = *(const v16bf*)(A  + (long long)srow * lda + kt * 32 + shalf);
        v16bf bv = *(const v16bf*)(Bt + (long long)srow * ldb + kt * 32 + shalf);
        *(v16bf*)(&sA[buf][srow * 32 + shalf]) = av;
        *(v16bf*)(&sB[buf][srow * 32 + shalf]) = bv;
        __syncthreads();
#endif

        v16bf bfrag[4];
#pragma unroll
        for (int nt = 0; nt < 4; ++nt)
            bfrag[nt] = *(const v16bf*)(&sB[buf][(wn * 64 + nt * 16 + fr) * 32 + kh]);

#pragma unroll
        for (int mt = 0; mt < 2; ++mt) {
            v16bf afrag = *(const v16bf*)(&sA[buf][(wm * 32 + mt * 16 + fr) * 32 + kh]);
#pragma unroll
            for (int nt = 0; nt < 4; ++nt) {
                acc[mt][nt] = __builtin_amdgcn_wmma_f32_16x16x32_bf16(
                    false, afrag, false, bfrag[nt],
                    (short)0, acc[mt][nt], false, false);
            }
        }
        __syncthreads();
    }

    // C/D layout: VGPR r: lanes 0-15 -> row r, lanes 16-31 -> row r+8; col = lane%16
    const int cn = lane & 15;
    const int rh = (lane >> 4) * 8;
#pragma unroll
    for (int mt = 0; mt < 2; ++mt) {
#pragma unroll
        for (int nt = 0; nt < 4; ++nt) {
            const int col  = n0 + wn * 64 + nt * 16 + cn;
            const float bb = bias ? bias[col] : 0.0f;
#pragma unroll
            for (int r = 0; r < 8; ++r) {
                const int row = m0 + wm * 32 + mt * 16 + rh + r;
                C[(long long)row * ldc + col] = acc[mt][nt][r] + bb;
            }
        }
    }
}

// ---------------------------------------------------------------------------
// Elementwise helpers (all tiny vs. the GEMMs)
// ---------------------------------------------------------------------------
__global__ void cvt_bf16_kernel(const float* __restrict__ src,
                                bf16_t* __restrict__ dst, int n) {
    int i = blockIdx.x * 256 + threadIdx.x;
    if (i < n) dst[i] = f2b(src[i]);
}

__global__ void exp_bf16_kernel(const float* __restrict__ src,
                                bf16_t* __restrict__ dst, int n) {
    int i = blockIdx.x * 256 + threadIdx.x;
    if (i < n) dst[i] = f2b(__expf(src[i]));
}

// wqkvT [768][1024] = concat(Wq,Wk,Wv)^T in bf16
__global__ void pack_wqkvT_kernel(const float* __restrict__ Wq,
                                  const float* __restrict__ Wk,
                                  const float* __restrict__ Wv,
                                  bf16_t* __restrict__ dst, int n) {
    int i = blockIdx.x * 256 + threadIdx.x;
    if (i >= n) return;
    int nr = i >> 10;          // 0..767 (output row = N index)
    int k  = i & 1023;
    float v = (nr < 256) ? Wq[k * 256 + nr]
            : (nr < 512) ? Wk[k * 256 + (nr - 256)]
                         : Wv[k * 256 + (nr - 512)];
    dst[i] = f2b(v);
}

__global__ void pack_bias_kernel(const float* __restrict__ bq,
                                 const float* __restrict__ bk,
                                 const float* __restrict__ bv,
                                 float* __restrict__ dst) {
    int j = blockIdx.x * 256 + threadIdx.x;
    if (j >= 768) return;
    dst[j] = (j < 256) ? bq[j] : (j < 512) ? bk[j - 256] : bv[j - 512];
}

// wpT [1024][256] = Wp[256][1024]^T in bf16
__global__ void cvt_wpT_kernel(const float* __restrict__ Wp,
                               bf16_t* __restrict__ dst, int n) {
    int i = blockIdx.x * 256 + threadIdx.x;
    if (i >= n) return;
    int nr = i >> 8;           // 0..1023
    int k  = i & 255;
    dst[i] = f2b(Wp[k * 1024 + nr]);
}

// From QKV f32 [M,768] build Zt bf16 [Bn][512][T]:
//   Zt[b][h][t] = exp(K)*V,  Zt[b][256+h][t] = exp(K)
__global__ void epi_ekv_kernel(const float* __restrict__ QKV,
                               bf16_t* __restrict__ Zt, int n) {
    int i = blockIdx.x * 256 + threadIdx.x;   // n = M*256
    if (i >= n) return;
    int row = i >> 8, h = i & 255;
    int b = row >> 11;          // / 2048
    int t = row & 2047;
    float kk = QKV[row * 768 + 256 + h];
    float vv = QKV[row * 768 + 512 + h];
    float ek = __expf(kk);
    long long base = (long long)b * 512 * 2048;
    Zt[base + (long long)h * 2048 + t]         = f2b(ek * vv);
    Zt[base + (long long)(256 + h) * 2048 + t] = f2b(ek);
}

// Yt bf16 [M,256] = sigmoid(Q) * num/den
__global__ void epi_gate_kernel(const float* __restrict__ QKV,
                                const float* __restrict__ ND,
                                bf16_t* __restrict__ Yt, int n) {
    int i = blockIdx.x * 256 + threadIdx.x;   // n = M*256
    if (i >= n) return;
    int row = i >> 8, h = i & 255;
    float q   = QKV[row * 768 + h];
    float sg  = 1.0f / (1.0f + __expf(-q));
    float num = ND[row * 512 + h];
    float den = ND[row * 512 + 256 + h];
    Yt[i] = f2b(sg * num / den);
}

// ---------------------------------------------------------------------------
// Launcher
// ---------------------------------------------------------------------------
extern "C" void kernel_launch(void* const* d_in, const int* in_sizes, int n_in,
                              void* d_out, int out_size, void* d_ws, size_t ws_size,
                              hipStream_t stream) {
    (void)in_sizes; (void)n_in; (void)out_size; (void)ws_size;

    const int Bn = 4, T = 2048, DIM = 1024, HID = 256;
    const int M  = Bn * T;         // 8192
    const int N1 = 3 * HID;        // 768

    const float* x     = (const float*)d_in[0];
    const float* Wq    = (const float*)d_in[1];
    const float* bq    = (const float*)d_in[2];
    const float* Wk    = (const float*)d_in[3];
    const float* bk    = (const float*)d_in[4];
    const float* Wv    = (const float*)d_in[5];
    const float* bv    = (const float*)d_in[6];
    const float* Wp    = (const float*)d_in[7];
    const float* bp    = (const float*)d_in[8];
    const float* wbias = (const float*)d_in[9];
    float* out = (float*)d_out;

    size_t off = 0;
    auto carve = [&](size_t bytes) {
        void* p = (char*)d_ws + off;
        off += (bytes + 255) & ~(size_t)255;
        return p;
    };
    bf16_t* xb    = (bf16_t*)carve((size_t)M * DIM * 2);       // 16 MB
    bf16_t* wqkvT = (bf16_t*)carve((size_t)N1 * DIM * 2);      // 1.5 MB
    float*  bqkv  = (float*) carve((size_t)N1 * 4);
    float*  qkv   = (float*) carve((size_t)M * N1 * 4);        // 24 MB
    bf16_t* ewb   = (bf16_t*)carve((size_t)T * T * 2);         // 8 MB
    bf16_t* zt    = (bf16_t*)carve((size_t)M * 2 * HID * 2);   // 8 MB
    float*  nd    = (float*) carve((size_t)M * 2 * HID * 4);   // 16 MB
    bf16_t* ytb   = (bf16_t*)carve((size_t)M * HID * 2);       // 4 MB
    bf16_t* wpT   = (bf16_t*)carve((size_t)DIM * HID * 2);     // 0.5 MB

    // 1) conversions / packing (B operands pre-transposed)
    {
        int n = M * DIM;
        cvt_bf16_kernel<<<(n + 255) / 256, 256, 0, stream>>>(x, xb, n);
    }
    {
        int n = N1 * DIM;
        pack_wqkvT_kernel<<<(n + 255) / 256, 256, 0, stream>>>(Wq, Wk, Wv, wqkvT, n);
        pack_bias_kernel<<<3, 256, 0, stream>>>(bq, bk, bv, bqkv);
    }
    {
        int n = T * T;
        exp_bf16_kernel<<<(n + 255) / 256, 256, 0, stream>>>(wbias, ewb, n);
    }
    {
        int n = DIM * HID;
        cvt_wpT_kernel<<<(n + 255) / 256, 256, 0, stream>>>(Wp, wpT, n);
    }

    // 2) fused QKV projection: qkv[M,768] = xb[M,1024] @ wqkvT[768,1024]^T + bqkv
    {
        dim3 g(N1 / 128, M / 128, 1);
        gemm_bf16f32<<<g, 256, 0, stream>>>(xb, DIM, 0LL,
                                            wqkvT, DIM, 0LL,
                                            qkv, N1, 0LL,
                                            bqkv, DIM);
    }

    // 3) Zt = [exp(K)*V | exp(K)]^T per batch
    {
        int n = M * HID;
        epi_ekv_kernel<<<(n + 255) / 256, 256, 0, stream>>>(qkv, zt, n);
    }

    // 4) fused num/den: nd[b][T,512] = ew[T,T] @ Zt[b][512,T]^T  (batched)
    {
        dim3 g((2 * HID) / 128, T / 128, Bn);
        gemm_bf16f32<<<g, 256, 0, stream>>>(ewb, T, 0LL,
                                            zt, T, (long long)2 * HID * T,
                                            nd, 2 * HID, (long long)T * 2 * HID,
                                            nullptr, T);
    }

    // 5) gate: Yt = sigmoid(Q) * num/den
    {
        int n = M * HID;
        epi_gate_kernel<<<(n + 255) / 256, 256, 0, stream>>>(qkv, nd, ytb, n);
    }

    // 6) output projection: out[M,1024] = Yt[M,256] @ wpT[1024,256]^T + bp
    {
        dim3 g(DIM / 128, M / 128, 1);
        gemm_bf16f32<<<g, 256, 0, stream>>>(ytb, HID, 0LL,
                                            wpT, HID, 0LL,
                                            out, DIM, 0LL,
                                            bp, HID);
    }
}
